// GNN_5360119186060
// MI455X (gfx1250) — compile-verified, matrix operand order checked by reading, not run on hardware
//
#include <hip/hip_runtime.h>

// ---------------------------------------------------------------------------
// GCN (4x GCNConv) for MI455X / gfx1250.
// GEMM: v_wmma_f32_16x16x32_f16, weights staged in LDS pre-swizzled into
//       B-fragment layout (one ds_load per fragment), activations loaded
//       with b128 and packed to f16 in-register.
// Aggregation: coalesced float4 gathers + hardware f32 global atomics
//       (destination rows are L2-resident: 25.6 MB << 192 MB L2).
// ---------------------------------------------------------------------------

typedef __attribute__((ext_vector_type(16))) _Float16 v16h;
typedef __attribute__((ext_vector_type(8)))  float    v8f;

#define THREADS 256

// ---------------- degree / norm ----------------
__global__ __launch_bounds__(THREADS) void k_zero(float* __restrict__ p, int n) {
  int i = blockIdx.x * blockDim.x + threadIdx.x;
  if (i < n) p[i] = 0.0f;
}

__global__ __launch_bounds__(THREADS) void k_deg_count(const int* __restrict__ dst,
                                                       float* __restrict__ deg, int E) {
  int e = blockIdx.x * blockDim.x + threadIdx.x;
  if (e < E) unsafeAtomicAdd(&deg[dst[e]], 1.0f);
}

__global__ __launch_bounds__(THREADS) void k_dis(float* __restrict__ deg, int n) {
  int i = blockIdx.x * blockDim.x + threadIdx.x;
  if (i < n) deg[i] = rsqrtf(deg[i] + 1.0f);  // +1 self loop; deg >= 1 always
}

// ---------------- dense GEMM: out[n,Fout] = act[n,64] @ W[64,Fout] ----------
// One wave per 16-row stripe; Fin fixed at 64 (two K=32 WMMA steps).
// W is staged once per block into LDS, pre-swizzled into per-lane 32-byte
// f16 B-fragments: fragment index = (colTile*2 + kstep)*32 + lane.
__global__ __launch_bounds__(THREADS) void k_gemm_wmma(const float* __restrict__ act,
                                                       const float* __restrict__ W,
                                                       float* __restrict__ out,
                                                       int nRows, int Fout, int lgF) {
  __shared__ v16h ldsB[4 * 2 * 32];              // max: 4 col tiles * 2 ksteps * 32 lanes (8 KB)
  _Float16* ldsH = (_Float16*)ldsB;

  const int tid = threadIdx.x;

  // ---- stage W -> LDS (coalesced b128 reads, f16 swizzled scatter) ----
  // element e: k = e>>lgF (K row), n = e&(Fout-1) (N col).
  // B-fragment addressing: c=n>>4, l2=n&15, ks=k>>5, kk=k&31,
  //   half=(kk>>3)&1, v = (kk&16)? 4+((kk&7)>>1) : (kk&7)>>1, pair=kk&1.
  {
    const int total = 64 * Fout;
    const float4* W4 = (const float4*)W;
    for (int e0 = tid * 16; e0 < total; e0 += THREADS * 16) {
      #pragma unroll
      for (int q = 0; q < 4; ++q) {
        float4 w = W4[(e0 >> 2) + q];
        float vals[4] = {w.x, w.y, w.z, w.w};
        #pragma unroll
        for (int j = 0; j < 4; ++j) {
          int e  = e0 + q * 4 + j;
          int k  = e >> lgF;
          int n  = e & (Fout - 1);
          int c  = n >> 4, l2 = n & 15;
          int ks = k >> 5, kk = k & 31;
          int half = (kk >> 3) & 1;
          int v  = (kk & 16) ? (4 + ((kk & 7) >> 1)) : ((kk & 7) >> 1);
          int pr = kk & 1;
          ldsH[(((c * 2 + ks) * 32) + (half * 16 + l2)) * 16 + 2 * v + pr] = (_Float16)vals[j];
        }
      }
    }
  }
  __syncthreads();

  const int wave    = (blockIdx.x * blockDim.x + tid) >> 5;
  const int rowBase = wave * 16;
  if (rowBase >= nRows) return;                  // wave-uniform: EXEC stays all-ones

  const int lane = tid & 31;
  const int l2   = lane & 15;
  const int half = lane >> 4;

  // ---- A fragments: per lane two contiguous 8-float runs per kstep (b128 x2) ----
  const float4* arow4 = (const float4*)(act + (size_t)(rowBase + l2) * 64);
  v16h afrag[2];
  #pragma unroll
  for (int ks = 0; ks < 2; ++ks) {
    int b0 = (ks * 32 + 8 * half) >> 2;          // K = ks*32 + 8*half + [0..7]
    int b1 = (ks * 32 + 16 + 8 * half) >> 2;     // K = ks*32 + 16 + 8*half + [0..7]
    float4 q0 = arow4[b0], q1 = arow4[b0 + 1];
    float4 q2 = arow4[b1], q3 = arow4[b1 + 1];
    afrag[ks][0]  = (_Float16)q0.x;  afrag[ks][1]  = (_Float16)q0.y;
    afrag[ks][2]  = (_Float16)q0.z;  afrag[ks][3]  = (_Float16)q0.w;
    afrag[ks][4]  = (_Float16)q1.x;  afrag[ks][5]  = (_Float16)q1.y;
    afrag[ks][6]  = (_Float16)q1.z;  afrag[ks][7]  = (_Float16)q1.w;
    afrag[ks][8]  = (_Float16)q2.x;  afrag[ks][9]  = (_Float16)q2.y;
    afrag[ks][10] = (_Float16)q2.z;  afrag[ks][11] = (_Float16)q2.w;
    afrag[ks][12] = (_Float16)q3.x;  afrag[ks][13] = (_Float16)q3.y;
    afrag[ks][14] = (_Float16)q3.z;  afrag[ks][15] = (_Float16)q3.w;
  }

  const int nColTiles = Fout >> 4;
  for (int c = 0; c < nColTiles; ++c) {
    v8f acc = {};
    #pragma unroll
    for (int ks = 0; ks < 2; ++ks) {
      v16h bfrag = ldsB[(c * 2 + ks) * 32 + lane];
      acc = __builtin_amdgcn_wmma_f32_16x16x32_f16(
          /*neg_a=*/false, afrag[ks], /*neg_b=*/false, bfrag,
          /*c_mod=*/(short)0, acc, /*reuse_a=*/false, /*reuse_b=*/false);
    }
    // C/D layout: VGPR r, lanes 0-15 -> M=r, lanes 16-31 -> M=8+r; N = l2.
    #pragma unroll
    for (int r = 0; r < 8; ++r) {
      int m = r + 8 * half;
      out[(size_t)(rowBase + m) * Fout + c * 16 + l2] = acc[r];
    }
  }
}

// ---------------- self-loop init: agg = msg * dis^2 ----------------
__global__ __launch_bounds__(THREADS) void k_self(const float* __restrict__ msg,
                                                  float* __restrict__ agg,
                                                  const float* __restrict__ dis,
                                                  int total, int lgF) {
  int i = blockIdx.x * blockDim.x + threadIdx.x;
  if (i >= total) return;
  float s = dis[i >> lgF];
  agg[i] = msg[i] * s * s;
}

// ---------------- edge scatter: agg[dst] += msg[src] * norm ----------------
// F/4 consecutive threads per edge (coalesced float4 row reads).
__global__ __launch_bounds__(THREADS) void k_scatter(const float* __restrict__ msg,
                                                     float* __restrict__ agg,
                                                     const float* __restrict__ dis,
                                                     const int* __restrict__ src,
                                                     const int* __restrict__ dst,
                                                     int E, int lgF4) {
  int tid = blockIdx.x * blockDim.x + threadIdx.x;
  int e = tid >> lgF4;
  if (e >= E) return;
  int c = tid & ((1 << lgF4) - 1);
  int s = src[e];
  int d = dst[e];
  float norm = dis[s] * dis[d];
  const float4 v = ((const float4*)msg)[((size_t)s << lgF4) + c];
  float* o = agg + ((((size_t)d << lgF4) + c) << 2);
  unsafeAtomicAdd(o + 0, v.x * norm);
  unsafeAtomicAdd(o + 1, v.y * norm);
  unsafeAtomicAdd(o + 2, v.z * norm);
  unsafeAtomicAdd(o + 3, v.w * norm);
}

// ---------------- bias (+ReLU) ----------------
__global__ __launch_bounds__(THREADS) void k_finalize(const float* __restrict__ agg,
                                                      const float* __restrict__ bias,
                                                      float* __restrict__ out,
                                                      int total, int maskF, int doRelu) {
  int i = blockIdx.x * blockDim.x + threadIdx.x;
  if (i >= total) return;
  float v = agg[i] + bias[i & maskF];
  out[i] = doRelu ? fmaxf(v, 0.0f) : v;
}

// ---------------------------------------------------------------------------
extern "C" void kernel_launch(void* const* d_in, const int* in_sizes, int n_in,
                              void* d_out, int out_size, void* d_ws, size_t ws_size,
                              hipStream_t stream) {
  const int N = 100000, HID = 64, OUTF = 16;
  const int E = in_sizes[1] / 2;

  const float* x     = (const float*)d_in[0];
  const int*   ei    = (const int*)d_in[1];
  const float* W_in  = (const float*)d_in[2];
  const float* b_in  = (const float*)d_in[3];
  const float* W_h   = (const float*)d_in[4];   // [2, 64, 64]
  const float* b_h   = (const float*)d_in[5];   // [2, 64]
  const float* W_out = (const float*)d_in[6];
  const float* b_out = (const float*)d_in[7];
  float* outp = (float*)d_out;

  const int* src = ei;
  const int* dst = ei + E;

  // workspace layout: dis[N] | G[N*64] | H[N*64]
  char* ws = (char*)d_ws;
  float* dis = (float*)ws;
  size_t off = (((size_t)N * 4) + 255) & ~(size_t)255;
  float* G = (float*)(ws + off);                 // gemm / message buffer
  float* H = G + (size_t)N * HID;                // activation / aggregation buffer

  auto blocks = [](long long n) { return (unsigned)((n + THREADS - 1) / THREADS); };

  // ---- deg_inv_sqrt ----
  k_zero<<<blocks(N), THREADS, 0, stream>>>(dis, N);
  k_deg_count<<<blocks(E), THREADS, 0, stream>>>(dst, dis, E);
  k_dis<<<blocks(N), THREADS, 0, stream>>>(dis, N);

  const int gemmThreads = (N / 16) * 32;         // one wave per 16-row tile

  // One GCNConv layer: G = act@W ; H = aggregate(G) ; act' = relu(H + b)
  auto layer = [&](const float* act, const float* W, const float* b,
                   float* outBuf, int Fout, int relu) {
    int lgF  = (Fout == 64) ? 6 : 4;
    int lgF4 = lgF - 2;
    k_gemm_wmma<<<blocks(gemmThreads), THREADS, 0, stream>>>(act, W, G, N, Fout, lgF);
    k_self<<<blocks((long long)N * Fout), THREADS, 0, stream>>>(G, H, dis, N * Fout, lgF);
    k_scatter<<<blocks((long long)E << lgF4), THREADS, 0, stream>>>(G, H, dis, src, dst, E, lgF4);
    k_finalize<<<blocks((long long)N * Fout), THREADS, 0, stream>>>(H, b, outBuf, N * Fout,
                                                                    Fout - 1, relu);
  };

  layer(x, W_in,              b_in,         H,    HID,  1);  // layer 1
  layer(H, W_h + 0 * 64 * 64, b_h + 0 * 64, H,    HID,  1);  // layer 2
  layer(H, W_h + 1 * 64 * 64, b_h + 1 * 64, H,    HID,  1);  // layer 3
  layer(H, W_out,             b_out,        outp, OUTF, 0);  // output layer (no ReLU)
}
